// GGNN_59425167507919
// MI455X (gfx1250) — matrix operand order repaired and checked by reading, not compiled, not run on hardware
//
#include <hip/hip_runtime.h>

// ---------------------------------------------------------------------------
// GGNN forward for MI455X (gfx1250, wave32, WMMA + Tensor Data Mover).
// GEMMs: v_wmma_f32_16x16x32_bf16 (f32 accumulate).
// A-tiles (activations) are DMA'd into LDS by the TDM (tensor_load_to_lds,
// TENSORcnt) with D# padding (row stride 132 floats) to avoid LDS bank
// conflicts on row-per-lane fragment reads.
// N=50000 (=16*3125), D=128, 3D=384, E=800000, 9 propagation iterations.
// ---------------------------------------------------------------------------

#define GG_N 50000
#define GG_E 800000
#define GG_D 128
#define GG_3D 384
#define GG_ITERS 9
#define GG_MBLK 3125   // N / 16
#define GG_LDSROW 132  // 128 + 4 DWORD TDM pad per row

typedef __attribute__((ext_vector_type(16))) __bf16 v16bf;
typedef __attribute__((ext_vector_type(8)))  __bf16 v8bf;
typedef __attribute__((ext_vector_type(8)))  float  v8f;
typedef __attribute__((ext_vector_type(4)))  unsigned int v4u;
typedef __attribute__((ext_vector_type(8)))  int v8i;
typedef __attribute__((ext_vector_type(4)))  int v4i;

// ---------------- helpers ----------------

__device__ __forceinline__ float sigmoidf_(float t) {
  return 1.0f / (1.0f + __expf(-t));
}

// TDM: DMA one [rows x 128] f32 tile (row stride 128 elems in memory) into LDS
// at lds_byte, padding 4 DWORDs after every 128 DWORDs (-> 132-float LDS rows).
// D# layout per CDNA5 ISA §8.3/8.4. 6-arg builtin form (clang-23 signature:
// g0 uint32x4, g1 int32x8, g2 int32x4, g3 int32x4, g4 int32x8, cpol).
__device__ __forceinline__ void tdm_load_tile_f32(const float* gsrc,
                                                  unsigned int lds_byte,
                                                  unsigned int rows) {
  const unsigned long long ga = (unsigned long long)(uintptr_t)gsrc;
  v4u g0;
  g0[0] = 1u;                                       // count=1 (valid), user mode
  g0[1] = lds_byte;                                 // lds_addr (bytes)
  g0[2] = (unsigned int)ga;                         // global_addr[31:0]
  g0[3] = (unsigned int)((ga >> 32) & 0x1FFFFFFu)   // global_addr[56:32]
        | (2u << 30);                               // type=2 ("image")
  v8i g1;
  g1[0] = (int)((2u << 16)        // data_size = 4 bytes
              | (1u << 20)        // pad_enable
              | (6u << 22)        // pad_interval: 2^(6+1)=128 DWORDs
              | (3u << 25));      // pad_amount: 3+1 = 4 DWORDs
  g1[1] = (int)((GG_D & 0xFFFFu) << 16);            // tensor_dim0[15:0]=128
  g1[2] = (int)(rows << 16);                        // dim0 hi=0 | tensor_dim1 lo=rows
  g1[3] = (int)(GG_D << 16);                        // dim1 hi=0 | tile_dim0=128
  g1[4] = (int)rows;                                // tile_dim1=rows | tile_dim2=0
  g1[5] = GG_D;                                     // tensor_dim0_stride[31:0]=128
  g1[6] = 0;
  g1[7] = 0;
  const v4i z4 = {0, 0, 0, 0};
  const v8i z8 = {0, 0, 0, 0, 0, 0, 0, 0};
  __builtin_amdgcn_tensor_load_to_lds(g0, g1, z4, z4, z8, 0);
}

// A fragment (16x32 bf16, M x K) from an LDS-staged f32 tile (row stride 132).
// Contiguous K order (matches B): lanes 0-15 -> M=lane, elems 0..15 = K 0..15;
//                 lanes 16-31 -> M=lane-16, elems 0..15 = K 16..31.
__device__ __forceinline__ v16bf load_a_frag(const float* __restrict__ row, int kb, int half) {
  const float4 a0 = *(const float4*)(row + kb + half * 16);
  const float4 a1 = *(const float4*)(row + kb + half * 16 + 4);
  const float4 a2 = *(const float4*)(row + kb + half * 16 + 8);
  const float4 a3 = *(const float4*)(row + kb + half * 16 + 12);
  v16bf f;
  f[0]  = (__bf16)a0.x; f[1]  = (__bf16)a0.y; f[2]  = (__bf16)a0.z; f[3]  = (__bf16)a0.w;
  f[4]  = (__bf16)a1.x; f[5]  = (__bf16)a1.y; f[6]  = (__bf16)a1.z; f[7]  = (__bf16)a1.w;
  f[8]  = (__bf16)a2.x; f[9]  = (__bf16)a2.y; f[10] = (__bf16)a2.z; f[11] = (__bf16)a2.w;
  f[12] = (__bf16)a3.x; f[13] = (__bf16)a3.y; f[14] = (__bf16)a3.z; f[15] = (__bf16)a3.w;
  return f;
}

// B fragment (32x16 bf16, K x N) from N-major bf16 weights Bt[Dout][128].
// Lane: N = lane&15, K-half = lane>>4; 16 contiguous bf16 along K.
__device__ __forceinline__ v16bf load_b_frag(const __bf16* __restrict__ brow, int kb) {
  const v8bf lo = *(const v8bf*)(brow + kb);
  const v8bf hi = *(const v8bf*)(brow + kb + 8);
  v16bf f;
#pragma unroll
  for (int i = 0; i < 8; ++i) { f[i] = lo[i]; f[i + 8] = hi[i]; }
  return f;
}

// ---------------- weight prep ----------------

__global__ void k_cvt_bf16(const float* __restrict__ src, __bf16* __restrict__ dst, int n) {
  int i = blockIdx.x * blockDim.x + threadIdx.x;
  if (i < n) dst[i] = (__bf16)src[i];
}

// gg_w[l] is used as x @ gg_w[l] (K-major). Store N-major: dst[l][j][k] = gg_w[l][k][j].
__global__ void k_cvt_gg_t(const float* __restrict__ gg, __bf16* __restrict__ dst) {
  int i = blockIdx.x * blockDim.x + threadIdx.x;     // 3*128*128 total
  if (i >= 3 * GG_D * GG_D) return;
  int l = i >> 14, r = i & 16383, j = r >> 7, k = r & 127;
  dst[i] = (__bf16)gg[l * GG_D * GG_D + k * GG_D + j];
}

__global__ void k_zero4(float* __restrict__ p, int n) {
  int i = blockIdx.x * blockDim.x + threadIdx.x;
  if (i * 4 < n) *(float4*)(p + i * 4) = make_float4(0.f, 0.f, 0.f, 0.f);
}

// ---------------- WMMA GEMM: C[M,Dout] = A[M,128] @ Bt^T + bias ----------------
// Bt is N-major bf16 [Dout][128]. Grid: (M/16, Dout/128); block: 256 (8 waves).
// A tile staged LDS via TDM once per block; all 8 waves read it from LDS.

__global__ __launch_bounds__(256) void k_gemm_bias(
    float* __restrict__ C, const float* __restrict__ A, const __bf16* __restrict__ Bt,
    const float* __restrict__ bias, int Dout)
{
  __shared__ __align__(16) float sh_a[16 * GG_LDSROW];

  const int lane = threadIdx.x & 31;
  const int wave = threadIdx.x >> 5;
  const int m0   = blockIdx.x * 16;
  const int n0   = (blockIdx.y * 8 + wave) * 16;
  const int half = lane >> 4;
  const int nm   = lane & 15;

  if (wave == 0) {
    tdm_load_tile_f32(A + (size_t)m0 * GG_D,
                      (unsigned int)(uintptr_t)(void*)sh_a, 16u);
    __builtin_amdgcn_s_wait_tensorcnt(0);
  }
  __syncthreads();

  const float*  arow = sh_a + nm * GG_LDSROW;
  const __bf16* brow = Bt + (size_t)(n0 + nm) * GG_D + half * 16;

  v8f acc = {};
#pragma unroll
  for (int ks = 0; ks < 4; ++ks) {
    const int kb = ks * 32;
    v16bf af = load_a_frag(arow, kb, half);
    v16bf bf = load_b_frag(brow, kb);
    acc = __builtin_amdgcn_wmma_f32_16x16x32_bf16(false, af, false, bf, (short)0, acc,
                                                  false, false);
  }
  const float bv = bias ? bias[n0 + nm] : 0.0f;
#pragma unroll
  for (int r = 0; r < 8; ++r)
    C[(size_t)(m0 + r + 8 * half) * Dout + n0 + nm] = acc[r] + bv;
}

// ---------------- edge scatter: agg[dst] += m[src] ----------------
// One thread per (edge, 4-float chunk): E*32 threads total.

__global__ __launch_bounds__(256) void k_scatter(
    float* __restrict__ agg, const float* __restrict__ m,
    const int* __restrict__ src, const int* __restrict__ dst)
{
  const int idx = blockIdx.x * 256 + threadIdx.x;      // < E*32
  const int e = idx >> 5;
  const int c = (idx & 31) * 4;
  const int s = src[e];
  const int d = dst[e];
  const float4 v = *(const float4*)(m + (size_t)s * GG_D + c);
  float* p = agg + (size_t)d * GG_D + c;
  atomicAdd(p + 0, v.x);
  atomicAdd(p + 1, v.y);
  atomicAdd(p + 2, v.z);
  atomicAdd(p + 3, v.w);
}

// ---------------- fused GRU cell ----------------
// xnew = GRU(agg, h). Each wave: one 16-col slice j0, 6 gate tiles, 24 WMMAs.
// agg and h 16x128 tiles staged into LDS by two TDM transfers per block.
// Wih_t/Whh_t are N-major bf16 [384][128] (== Wih/Whh row-major, ref uses W.T).

__global__ __launch_bounds__(256) void k_gru(
    float* __restrict__ xnew, const float* __restrict__ agg, const float* __restrict__ h,
    const __bf16* __restrict__ Wih_t, const __bf16* __restrict__ Whh_t,
    const float* __restrict__ bih, const float* __restrict__ bhh)
{
  __shared__ __align__(16) float sh_a[16 * GG_LDSROW];
  __shared__ __align__(16) float sh_h[16 * GG_LDSROW];

  const int lane = threadIdx.x & 31;
  const int wave = threadIdx.x >> 5;
  const int m0   = blockIdx.x * 16;
  const int j0   = wave * 16;
  const int half = lane >> 4;
  const int nm   = lane & 15;

  if (wave == 0) {
    tdm_load_tile_f32(agg + (size_t)m0 * GG_D,
                      (unsigned int)(uintptr_t)(void*)sh_a, 16u);
    tdm_load_tile_f32(h + (size_t)m0 * GG_D,
                      (unsigned int)(uintptr_t)(void*)sh_h, 16u);
    __builtin_amdgcn_s_wait_tensorcnt(0);
  }
  __syncthreads();

  const float* aggrow = sh_a + nm * GG_LDSROW;
  const float* hrow   = sh_h + nm * GG_LDSROW;

  const __bf16* bir_p = Wih_t + (size_t)(j0 + nm) * GG_D + half * 16;
  const __bf16* biz_p = Wih_t + (size_t)(128 + j0 + nm) * GG_D + half * 16;
  const __bf16* bin_p = Wih_t + (size_t)(256 + j0 + nm) * GG_D + half * 16;
  const __bf16* bhr_p = Whh_t + (size_t)(j0 + nm) * GG_D + half * 16;
  const __bf16* bhz_p = Whh_t + (size_t)(128 + j0 + nm) * GG_D + half * 16;
  const __bf16* bhn_p = Whh_t + (size_t)(256 + j0 + nm) * GG_D + half * 16;

  v8f air = {}, aiz = {}, ain = {}, ahr = {}, ahz = {}, ahn = {};
#pragma unroll
  for (int ks = 0; ks < 4; ++ks) {
    const int kb = ks * 32;
    v16bf af = load_a_frag(aggrow, kb, half);
    v16bf hf = load_a_frag(hrow, kb, half);
    air = __builtin_amdgcn_wmma_f32_16x16x32_bf16(false, af, false, load_b_frag(bir_p, kb),
                                                  (short)0, air, false, false);
    aiz = __builtin_amdgcn_wmma_f32_16x16x32_bf16(false, af, false, load_b_frag(biz_p, kb),
                                                  (short)0, aiz, false, false);
    ain = __builtin_amdgcn_wmma_f32_16x16x32_bf16(false, af, false, load_b_frag(bin_p, kb),
                                                  (short)0, ain, false, false);
    ahr = __builtin_amdgcn_wmma_f32_16x16x32_bf16(false, hf, false, load_b_frag(bhr_p, kb),
                                                  (short)0, ahr, false, false);
    ahz = __builtin_amdgcn_wmma_f32_16x16x32_bf16(false, hf, false, load_b_frag(bhz_p, kb),
                                                  (short)0, ahz, false, false);
    ahn = __builtin_amdgcn_wmma_f32_16x16x32_bf16(false, hf, false, load_b_frag(bhn_p, kb),
                                                  (short)0, ahn, false, false);
  }

  const float c_ir = bih[j0 + nm],       c_hr = bhh[j0 + nm];
  const float c_iz = bih[128 + j0 + nm], c_hz = bhh[128 + j0 + nm];
  const float c_in = bih[256 + j0 + nm], c_hn = bhh[256 + j0 + nm];

#pragma unroll
  for (int r = 0; r < 8; ++r) {
    const int row = m0 + r + 8 * half;
    const float rg = sigmoidf_((air[r] + c_ir) + (ahr[r] + c_hr));
    const float zg = sigmoidf_((aiz[r] + c_iz) + (ahz[r] + c_hz));
    const float ng = tanhf((ain[r] + c_in) + rg * (ahn[r] + c_hn));
    const float hv = sh_h[(r + 8 * half) * GG_LDSROW + j0 + nm];  // staged h reuse
    xnew[(size_t)row * GG_D + j0 + nm] = (1.0f - zg) * ng + zg * hv;
  }
}

// ---------------- pooling + output ----------------

__global__ void k_colsum(float* __restrict__ pooled, const float* __restrict__ x) {
  const int d = threadIdx.x;                 // 128 threads
  float s = 0.0f;
  for (int r = blockIdx.x; r < GG_N; r += gridDim.x)
    s += x[(size_t)r * GG_D + d];
  atomicAdd(&pooled[d], s);
}

__global__ void k_final(float* __restrict__ out, const float* __restrict__ pooled,
                        const float* __restrict__ W_out, const float* __restrict__ b_out) {
  const int o = threadIdx.x;
  if (o < 2) {
    float s = b_out[o];
    const float inv_n = 1.0f / (float)GG_N;
    for (int d = 0; d < GG_D; ++d)
      s += (pooled[d] * inv_n) * W_out[o * GG_D + d];
    out[o] = s;
  }
}

// ---------------- launch ----------------

extern "C" void kernel_launch(void* const* d_in, const int* in_sizes, int n_in,
                              void* d_out, int out_size, void* d_ws, size_t ws_size,
                              hipStream_t stream) {
  (void)in_sizes; (void)n_in; (void)out_size; (void)ws_size;

  const float* node_embed = (const float*)d_in[0];   // [N, D]
  const int*   edges      = (const int*)d_in[1];     // [2, E]
  const float* W_in       = (const float*)d_in[2];   // [D, D]
  const float* b_in       = (const float*)d_in[3];   // [D]
  const float* gg_w       = (const float*)d_in[4];   // [3, D, D]
  const float* Wih        = (const float*)d_in[5];   // [3D, D]
  const float* Whh        = (const float*)d_in[6];   // [3D, D]
  const float* bih        = (const float*)d_in[7];   // [3D]
  const float* bhh        = (const float*)d_in[8];   // [3D]
  const float* W_out      = (const float*)d_in[9];   // [2, D]
  const float* b_out      = (const float*)d_in[10];  // [2]

  const int* src = edges;
  const int* dst = edges + GG_E;

  const size_t ND = (size_t)GG_N * GG_D;             // 6.4M floats
  float* xa     = (float*)d_ws;
  float* xb     = xa + ND;
  float* mbuf   = xb + ND;
  float* aggb   = mbuf + ND;
  float* pooled = aggb + ND;                         // 128 floats (pad to 256)
  __bf16* wWin  = (__bf16*)(pooled + 256);           // 128*128
  __bf16* wWg   = wWin + GG_D * GG_D;                // 3*128*128
  __bf16* wWih  = wWg + 3 * GG_D * GG_D;             // 384*128
  __bf16* wWhh  = wWih + GG_3D * GG_D;               // 384*128

  // Weight prep (bf16). W_in/Wih/Whh are applied as W.T -> row-major == N-major.
  k_cvt_bf16<<<(GG_D * GG_D + 255) / 256, 256, 0, stream>>>(W_in, wWin, GG_D * GG_D);
  k_cvt_bf16<<<(GG_3D * GG_D + 255) / 256, 256, 0, stream>>>(Wih, wWih, GG_3D * GG_D);
  k_cvt_bf16<<<(GG_3D * GG_D + 255) / 256, 256, 0, stream>>>(Whh, wWhh, GG_3D * GG_D);
  k_cvt_gg_t<<<(3 * GG_D * GG_D + 255) / 256, 256, 0, stream>>>(gg_w, wWg);

  // x = node_embed @ W_in.T + b_in
  k_gemm_bias<<<dim3(GG_MBLK, 1), 256, 0, stream>>>(xa, node_embed, wWin, b_in, GG_D);

  float* xc = xa;
  float* xn = xb;
  for (int it = 0; it < GG_ITERS; ++it) {
    const int l = it % 3;
    // m = x @ gg_w[l]
    k_gemm_bias<<<dim3(GG_MBLK, 1), 256, 0, stream>>>(mbuf, xc, wWg + (size_t)l * GG_D * GG_D,
                                                      (const float*)nullptr, GG_D);
    // agg = segment_sum(m[src], dst)
    k_zero4<<<(int)((ND / 4 + 255) / 256), 256, 0, stream>>>(aggb, (int)ND);
    k_scatter<<<GG_E * 32 / 256, 256, 0, stream>>>(aggb, mbuf, src, dst);
    // x = GRU(agg, x)
    k_gru<<<GG_MBLK, 256, 0, stream>>>(xn, aggb, xc, wWih, wWhh, bih, bhh);
    float* t = xc; xc = xn; xn = t;
  }

  // mean pool + output linear
  k_zero4<<<1, 64, 0, stream>>>(pooled, GG_D);
  k_colsum<<<250, GG_D, 0, stream>>>(pooled, xc);
  k_final<<<1, 32, 0, stream>>>((float*)d_out, pooled, W_out, b_out);
}
